// Net_64982855188851
// MI455X (gfx1250) — matrix operand order repaired and checked by reading, not compiled, not run on hardware
//
#include <hip/hip_runtime.h>
#include <hip/hip_bf16.h>
#include <math.h>

typedef _Float16 v16h __attribute__((ext_vector_type(16)));
typedef _Float16 v8h  __attribute__((ext_vector_type(8)));
typedef float    v8f  __attribute__((ext_vector_type(8)));

#define EPS_BN 1e-5f

static __device__ __forceinline__ float lrelu_f(float x) { return x >= 0.f ? x : 0.2f * x; }

// ---------------------------------------------------------------------------
// Device kernels
// ---------------------------------------------------------------------------

// Convert f32 (R x C, leading dim lds) into zero-padded f16 (Rp x Cp).
__global__ void k_cvt_pad_f16(const float* __restrict__ src, int R, int C, int lds,
                              _Float16* __restrict__ dst, int Cp, long total)
{
    long i = (long)blockIdx.x * blockDim.x + threadIdx.x;
    if (i >= total) return;
    int r = (int)(i / Cp), c = (int)(i % Cp);
    float v = (r < R && c < C) ? src[(long)r * lds + c] : 0.f;
    dst[i] = (_Float16)v;
}

// C = A(f16, R x Kp) @ B(f16, Kp x Cp) + bias, fp32 accumulate via WMMA.
// One wave per 16x16 output tile. Kp % 32 == 0, ldc % 16 == 0, R % 16 == 0.
__global__ void k_gemm_wmma(const _Float16* __restrict__ A, const _Float16* __restrict__ B,
                            const float* __restrict__ bias, float* __restrict__ Cm,
                            int R, int Cc, int Kp, int lda, int ldb, int ldc)
{
    const int lane = threadIdx.x;                 // 0..31, wave32
    const int tn = blockIdx.x, tm = blockIdx.y;
    const int mrow = tm * 16 + (lane & 15);       // A row for this lane
    const int kb   = (lane >> 4) * 8;             // K sub-block per lane half
    const _Float16* Arow = A + (long)mrow * lda;
    v8f acc = {};
    for (int k0 = 0; k0 < Kp; k0 += 32) {
        v8h alo = *(const v8h*)(Arow + k0 + kb);
        v8h ahi = *(const v8h*)(Arow + k0 + kb + 16);
        v16h a = __builtin_shufflevector(alo, ahi, 0,1,2,3,4,5,6,7,8,9,10,11,12,13,14,15);
        const _Float16* Bp = B + (long)(k0 + kb) * ldb + tn * 16 + (lane & 15);
        v16h b;
#pragma unroll
        for (int s = 0; s < 16; ++s) b[s] = Bp[(long)(s < 8 ? s : 8 + s) * ldb];
        acc = __builtin_amdgcn_wmma_f32_16x16x32_f16(false, a, false, b, (short)0, acc,
                                                     false, false);
    }
    const int col   = tn * 16 + (lane & 15);
    const int rbase = tm * 16 + ((lane >> 4) ? 8 : 0);
    float bv = (bias && col < Cc) ? bias[col] : 0.f;
#pragma unroll
    for (int r = 0; r < 8; ++r) {
        float v = acc[r] + bv;
        if (col >= Cc) v = 0.f;                   // keep padding columns zeroed
        Cm[(long)(rbase + r) * ldc + col] = v;
    }
}

// Per-channel mean/var over R rows (one block per channel).
__global__ void k_bn_stats(const float* __restrict__ x, int R, int ld,
                           float* __restrict__ mean, float* __restrict__ var)
{
    __shared__ float ss[256], ss2[256];
    const int c = blockIdx.x, t = threadIdx.x;
    float s = 0.f, s2 = 0.f;
    for (int r = t; r < R; r += 256) {
        float v = x[(long)r * ld + c];
        s += v; s2 += v * v;
    }
    ss[t] = s; ss2[t] = s2;
    __syncthreads();
    for (int o = 128; o > 0; o >>= 1) {
        if (t < o) { ss[t] += ss[t + o]; ss2[t] += ss2[t + o]; }
        __syncthreads();
    }
    if (t == 0) {
        float m = ss[0] / (float)R;
        mean[c] = m;
        var[c]  = ss2[0] / (float)R - m * m;
    }
}

__global__ void k_bn_apply(float* __restrict__ x, int R, int C, int ld,
                           const float* __restrict__ mean, const float* __restrict__ var,
                           const float* __restrict__ g, const float* __restrict__ bt,
                           int act, long total)
{
    long i = (long)blockIdx.x * blockDim.x + threadIdx.x;
    if (i >= total) return;
    int r = (int)(i / C), c = (int)(i % C);
    float v = x[(long)r * ld + c];
    v = (v - mean[c]) * rsqrtf(var[c] + EPS_BN) * g[c] + bt[c];
    if (act) v = lrelu_f(v);
    x[(long)r * ld + c] = v;
}

// Brute-force KNN top-16 (ascending d2). LDS-tiled source points,
// fully unrolled register-resident sorted list (static indexing only).
__global__ void k_knn16(const float* __restrict__ q, int nq,
                        const float* __restrict__ s, int ns, int* __restrict__ out)
{
    __shared__ float sx[256], sy[256], sz[256];
    const int t = threadIdx.x;
    const int qi = blockIdx.x * blockDim.x + t;
    float qx = 0.f, qy = 0.f, qz = 0.f;
    if (qi < nq) { qx = q[qi * 3]; qy = q[qi * 3 + 1]; qz = q[qi * 3 + 2]; }
    float bd[16]; int bi[16];
#pragma unroll
    for (int u = 0; u < 16; ++u) { bd[u] = 3.4e38f; bi[u] = 0; }
    for (int s0 = 0; s0 < ns; s0 += 256) {
        int m = ns - s0; if (m > 256) m = 256;
        if (t < m) {
            sx[t] = s[(s0 + t) * 3];
            sy[t] = s[(s0 + t) * 3 + 1];
            sz[t] = s[(s0 + t) * 3 + 2];
        }
        __syncthreads();
        if (qi < nq) {
            if (s0 + 256 < ns) __builtin_prefetch(s + (size_t)(s0 + 256) * 3, 0, 0);
            for (int jj = 0; jj < m; ++jj) {
                float dx = qx - sx[jj], dy = qy - sy[jj], dz = qz - sz[jj];
                float d2 = dx * dx + dy * dy + dz * dz;
                if (d2 < bd[15]) {
                    float cd = d2; int ci = s0 + jj;
#pragma unroll
                    for (int u = 0; u < 16; ++u) {
                        if (cd < bd[u]) {
                            float td = bd[u]; int ti = bi[u];
                            bd[u] = cd; bi[u] = ci;
                            cd = td; ci = ti;
                        }
                    }
                }
            }
        }
        __syncthreads();
    }
    if (qi < nq) {
#pragma unroll
        for (int u = 0; u < 16; ++u) out[qi * 16 + u] = bi[u];
    }
}

// Nearest-neighbor (k=1).
__global__ void k_nn1(const float* __restrict__ q, int nq,
                      const float* __restrict__ s, int ns, int* __restrict__ out)
{
    __shared__ float sx[256], sy[256], sz[256];
    const int t = threadIdx.x;
    const int qi = blockIdx.x * blockDim.x + t;
    float qx = 0.f, qy = 0.f, qz = 0.f;
    if (qi < nq) { qx = q[qi * 3]; qy = q[qi * 3 + 1]; qz = q[qi * 3 + 2]; }
    float bd = 3.4e38f; int bi = 0;
    for (int s0 = 0; s0 < ns; s0 += 256) {
        int m = ns - s0; if (m > 256) m = 256;
        if (t < m) {
            sx[t] = s[(s0 + t) * 3];
            sy[t] = s[(s0 + t) * 3 + 1];
            sz[t] = s[(s0 + t) * 3 + 2];
        }
        __syncthreads();
        if (qi < nq) {
            for (int jj = 0; jj < m; ++jj) {
                float dx = qx - sx[jj], dy = qy - sy[jj], dz = qz - sz[jj];
                float d2 = dx * dx + dy * dy + dz * dz;
                if (d2 < bd) { bd = d2; bi = s0 + jj; }
            }
        }
        __syncthreads();
    }
    if (qi < nq) out[qi] = bi;
}

// Build relative-position encoding rel = [pi, pj, pi-pj, |pi-pj|] directly as f16
// padded to K=32 (GEMM A operand).
__global__ void k_build_rel16(const float* __restrict__ pos, const int* __restrict__ idx,
                              int N, _Float16* __restrict__ rel)
{
    int i = blockIdx.x * blockDim.x + threadIdx.x;
    if (i >= N * 16) return;
    int n = i >> 4;
    int j = idx[i];
    float pix = pos[n * 3], piy = pos[n * 3 + 1], piz = pos[n * 3 + 2];
    float pjx = pos[j * 3], pjy = pos[j * 3 + 1], pjz = pos[j * 3 + 2];
    float dx = pix - pjx, dy = piy - pjy, dz = piz - pjz;
    float d = sqrtf(dx * dx + dy * dy + dz * dz + 1e-12f);
    _Float16* r = rel + (long)i * 32;
    r[0] = (_Float16)pix; r[1] = (_Float16)piy; r[2] = (_Float16)piz;
    r[3] = (_Float16)pjx; r[4] = (_Float16)pjy; r[5] = (_Float16)pjz;
    r[6] = (_Float16)dx;  r[7] = (_Float16)dy;  r[8] = (_Float16)dz;
    r[9] = (_Float16)d;
#pragma unroll
    for (int u = 10; u < 32; ++u) r[u] = (_Float16)0.f;
}

// local = concat(x[idx], enc), emitted both as f32 (for softmax reduce) and f16 (GEMM A).
__global__ void k_gather_local(const float* __restrict__ x, int ldx, int cin,
                               const float* __restrict__ enc, int lde, int cenc,
                               const int* __restrict__ idx, int N,
                               float* __restrict__ loc, int ldl,
                               _Float16* __restrict__ loc16, int ldl16)
{
    int row = blockIdx.x * blockDim.x + threadIdx.x;
    if (row >= N * 16) return;
    int j = idx[row];
    const float* xs = x + (long)j * ldx;
    const float* es = enc + (long)row * lde;
    float* o = loc + (long)row * ldl;
    _Float16* o16 = loc16 + (long)row * ldl16;
    int c = 0;
    for (int u = 0; u < cin; ++u, ++c)  { float v = xs[u]; o[c] = v; o16[c] = (_Float16)v; }
    for (int u = 0; u < cenc; ++u, ++c) { float v = es[u]; o[c] = v; o16[c] = (_Float16)v; }
    for (int u = c; u < ldl; ++u)  o[u] = 0.f;
    for (int u = c; u < ldl16; ++u) o16[u] = (_Float16)0.f;
}

// Softmax over the 16 neighbors (per point, per channel) + weighted sum of local.
__global__ void k_attn(const float* __restrict__ att, int lda,
                       const float* __restrict__ loc, int ldl,
                       int N, int C, float* __restrict__ out, int ldo, long total)
{
    long i = (long)blockIdx.x * blockDim.x + threadIdx.x;
    if (i >= total) return;
    int n = (int)(i / ldo), c = (int)(i % ldo);
    if (c >= C) { out[i] = 0.f; return; }
    const float* ap = att + (long)n * 16 * lda + c;
    const float* lp = loc + (long)n * 16 * ldl + c;
    float mx = -3.4e38f;
#pragma unroll
    for (int k = 0; k < 16; ++k) { float v = ap[(long)k * lda]; mx = v > mx ? v : mx; }
    float den = 0.f, num = 0.f;
#pragma unroll
    for (int k = 0; k < 16; ++k) {
        float e = expf(ap[(long)k * lda] - mx);
        den += e;
        num += e * lp[(long)k * ldl];
    }
    out[i] = num / den;
}

// out = concat(xa[nn[r]], xb[r]) with zero-padding to ldo.
__global__ void k_concat_nn(const float* __restrict__ xa, int lda_, int ca,
                            const int* __restrict__ nn,
                            const float* __restrict__ xb, int ldb_, int cb,
                            int N, float* __restrict__ out, int ldo)
{
    int r = blockIdx.x * blockDim.x + threadIdx.x;
    if (r >= N) return;
    int j = nn[r];
    const float* pa = xa + (long)j * lda_;
    const float* pb = xb + (long)r * ldb_;
    float* o = out + (long)r * ldo;
    int c = 0;
    for (int u = 0; u < ca; ++u, ++c) o[c] = pa[u];
    for (int u = 0; u < cb; ++u, ++c) o[c] = pb[u];
    for (; c < ldo; ++c) o[c] = 0.f;
}

__global__ void k_add_lrelu(const float* __restrict__ a, const float* __restrict__ b,
                            float* __restrict__ o, long total)
{
    long i = (long)blockIdx.x * blockDim.x + threadIdx.x;
    if (i >= total) return;
    o[i] = lrelu_f(a[i] + b[i]);
}

__global__ void k_logsoftmax(const float* __restrict__ logits, int ld, int N, int C,
                             float* __restrict__ out)
{
    int r = blockIdx.x * blockDim.x + threadIdx.x;
    if (r >= N) return;
    const float* p = logits + (long)r * ld;
    float mx = -3.4e38f;
    for (int c = 0; c < C; ++c) mx = p[c] > mx ? p[c] : mx;
    float s = 0.f;
    for (int c = 0; c < C; ++c) s += expf(p[c] - mx);
    float lse = mx + logf(s);
    for (int c = 0; c < C; ++c) out[(long)r * C + c] = p[c] - lse;
}

// ---------------------------------------------------------------------------
// Host-side orchestration
// ---------------------------------------------------------------------------

static inline int rup(int x, int m) { return (x + m - 1) / m * m; }

struct Arena { char* base; size_t off; size_t cap; };
static inline void* aalloc(Arena& a, size_t bytes)
{
    size_t o = (a.off + 255) & ~(size_t)255;
    a.off = o + bytes;
    return a.base + o;
}

struct Lin  { const float *W, *b, *g, *bt; int din, dout; };
struct LfaP { Lin enc; const float* attW; Lin post; int c; };
struct BlockP { Lin mlp1, shortcut, mlp2; LfaP l1, l2; };

static void h_cvt16(hipStream_t st, const float* src, int R, int C, int lds,
                    _Float16* dst, int Rp, int Cp)
{
    long total = (long)Rp * Cp;
    k_cvt_pad_f16<<<(unsigned)((total + 255) / 256), 256, 0, st>>>(src, R, C, lds, dst, Cp, total);
}

static void h_gemm(hipStream_t st, const _Float16* A, const _Float16* B, const float* bias,
                   float* C, int R, int Cc, int Kp, int lda, int ldb, int ldc)
{
    dim3 grid(ldc / 16, R / 16);
    k_gemm_wmma<<<grid, 32, 0, st>>>(A, B, bias, C, R, Cc, Kp, lda, ldb, ldc);
}

static void h_bn(hipStream_t st, Arena scratch, float* x, int R, int C, int ld,
                 const float* g, const float* bt, int act)
{
    float* mean = (float*)aalloc(scratch, (size_t)C * 4);
    float* var  = (float*)aalloc(scratch, (size_t)C * 4);
    k_bn_stats<<<C, 256, 0, st>>>(x, R, ld, mean, var);
    long tot = (long)R * C;
    k_bn_apply<<<(unsigned)((tot + 255) / 256), 256, 0, st>>>(x, R, C, ld, mean, var, g, bt,
                                                              act, tot);
}

// out (R x rup16(dout) f32) = optional-BN/act( X @ W + b )
static void h_lin(hipStream_t st, Arena scratch, const float* X, int R, int K, int ldx,
                  const Lin& L, float* out, int act, int use_bn)
{
    const int Kp = rup(K, 32), Cp = rup(L.dout, 16);
    _Float16* A16 = (_Float16*)aalloc(scratch, (size_t)R * Kp * 2);
    h_cvt16(st, X, R, K, ldx, A16, R, Kp);
    _Float16* W16 = (_Float16*)aalloc(scratch, (size_t)Kp * Cp * 2);
    h_cvt16(st, L.W, K, L.dout, L.dout, W16, Kp, Cp);
    h_gemm(st, A16, W16, L.b, out, R, L.dout, Kp, Kp, Cp, Cp);
    if (use_bn) h_bn(st, scratch, out, R, L.dout, Cp, L.g, L.bt, act);
}

// Local Feature Aggregation. x: N x ld(cin), out: N x rup16(c).
static void h_lfa(hipStream_t st, Arena scratch, const float* x, int ldx,
                  const float* pos, const int* idx, int N, const LfaP& P, float* out)
{
    const int c = P.c, cin = c / 2, cenc = c / 2;
    const int N16 = N * 16;
    const int encCp = rup(cenc, 16);
    float* enc = (float*)aalloc(scratch, (size_t)N16 * encCp * 4);
    {   // rel encoding + enc = lin_bn_act(rel)
        Arena sub = scratch;
        _Float16* rel16 = (_Float16*)aalloc(sub, (size_t)N16 * 32 * 2);
        k_build_rel16<<<(N16 + 255) / 256, 256, 0, st>>>(pos, idx, N, rel16);
        _Float16* W16 = (_Float16*)aalloc(sub, (size_t)32 * encCp * 2);
        h_cvt16(st, P.enc.W, 10, cenc, cenc, W16, 32, encCp);
        h_gemm(st, rel16, W16, P.enc.b, enc, N16, cenc, 32, 32, encCp, encCp);
        h_bn(st, sub, enc, N16, cenc, encCp, P.enc.g, P.enc.bt, 1);
    }
    const int Cp = rup(c, 16), Kp = rup(c, 32);
    float* local = (float*)aalloc(scratch, (size_t)N16 * Cp * 4);
    float* att   = (float*)aalloc(scratch, (size_t)N16 * Cp * 4);
    {   // att = local @ att_W
        Arena sub = scratch;
        _Float16* local16 = (_Float16*)aalloc(sub, (size_t)N16 * Kp * 2);
        k_gather_local<<<(N16 + 255) / 256, 256, 0, st>>>(x, ldx, cin, enc, encCp, cenc,
                                                          idx, N, local, Cp, local16, Kp);
        _Float16* aW16 = (_Float16*)aalloc(sub, (size_t)Kp * Cp * 2);
        h_cvt16(st, P.attW, c, c, c, aW16, Kp, Cp);
        h_gemm(st, local16, aW16, nullptr, att, N16, c, Kp, Kp, Cp, Cp);
    }
    float* osum = (float*)aalloc(scratch, (size_t)N * Cp * 4);
    long tot = (long)N * Cp;
    k_attn<<<(unsigned)((tot + 255) / 256), 256, 0, st>>>(att, Cp, local, Cp, N, c,
                                                          osum, Cp, tot);
    h_lin(st, scratch, osum, N, c, Cp, P.post, out, 1, 1);
}

static void h_block(hipStream_t st, Arena scratch, const float* x, int ldx, int N,
                    int din, int dout, const BlockP& B, const float* pos, float* out)
{
    int* idx = (int*)aalloc(scratch, (size_t)N * 16 * 4);
    k_knn16<<<(N + 255) / 256, 256, 0, st>>>(pos, N, pos, N, idx);
    const int d8 = dout / 8, c1 = dout / 4, c2 = dout / 2;
    const int Cp = rup(dout, 16);
    float* sc = (float*)aalloc(scratch, (size_t)N * Cp * 4);
    h_lin(st, scratch, x, N, din, ldx, B.shortcut, sc, 0, 1);
    const int d8p = rup(d8, 16);
    float* h = (float*)aalloc(scratch, (size_t)N * d8p * 4);
    h_lin(st, scratch, x, N, din, ldx, B.mlp1, h, 1, 1);
    const int c1p = rup(c1, 16);
    float* o1 = (float*)aalloc(scratch, (size_t)N * c1p * 4);
    h_lfa(st, scratch, h, d8p, pos, idx, N, B.l1, o1);
    const int c2p = rup(c2, 16);
    float* o2 = (float*)aalloc(scratch, (size_t)N * c2p * 4);
    h_lfa(st, scratch, o1, c1p, pos, idx, N, B.l2, o2);
    float* m2 = (float*)aalloc(scratch, (size_t)N * Cp * 4);
    h_lin(st, scratch, o2, N, c2, c2p, B.mlp2, m2, 0, 1);
    long tot = (long)N * Cp;
    k_add_lrelu<<<(unsigned)((tot + 255) / 256), 256, 0, st>>>(m2, sc, out, tot);
}

// Feature-propagation: out = lin_bn_act(concat(x[nn1(pos_skip, pos)], x_skip))
static void h_fp(hipStream_t st, Arena scratch,
                 const float* x, int ldx, int cx, const float* pos, int Nsrc,
                 const float* xs, int ldxs, int cs, const float* poss, int Ns,
                 const Lin& L, float* out)
{
    int* nn = (int*)aalloc(scratch, (size_t)Ns * 4);
    k_nn1<<<(Ns + 255) / 256, 256, 0, st>>>(poss, Ns, pos, Nsrc, nn);
    const int cc = cx + cs, ccp = rup(cc, 16);
    float* h = (float*)aalloc(scratch, (size_t)Ns * ccp * 4);
    k_concat_nn<<<(Ns + 255) / 256, 256, 0, st>>>(x, ldx, cx, nn, xs, ldxs, cs, Ns, h, ccp);
    h_lin(st, scratch, h, Ns, cc, ccp, L, out, 1, 1);
}

// ---------------------------------------------------------------------------

extern "C" void kernel_launch(void* const* d_in, const int* in_sizes, int n_in,
                              void* d_out, int out_size, void* d_ws, size_t ws_size,
                              hipStream_t stream)
{
    (void)in_sizes; (void)n_in; (void)out_size;
    const float* x   = (const float*)d_in[0];
    const float* pos = (const float*)d_in[1];
    int cur = 2;

    auto getLin = [&](int din, int dout, bool norm) -> Lin {
        Lin L; L.din = din; L.dout = dout;
        L.W = (const float*)d_in[cur++];
        L.b = (const float*)d_in[cur++];
        if (norm) { L.g = (const float*)d_in[cur++]; L.bt = (const float*)d_in[cur++]; }
        else      { L.g = nullptr; L.bt = nullptr; }
        return L;
    };
    auto getLfa = [&](int c) -> LfaP {
        LfaP P; P.c = c;
        P.enc  = getLin(10, c / 2, true);
        P.attW = (const float*)d_in[cur++];
        P.post = getLin(c, c, true);
        return P;
    };
    auto getBlock = [&](int din, int dout) -> BlockP {
        BlockP B;
        B.mlp1     = getLin(din, dout / 8, true);
        B.shortcut = getLin(din, dout, true);
        B.mlp2     = getLin(dout / 2, dout, true);
        B.l1       = getLfa(dout / 4);
        B.l2       = getLfa(dout / 2);
        return B;
    };

    Lin fc0 = getLin(6, 32, false);
    BlockP b1 = getBlock(32, 32);
    BlockP b2 = getBlock(32, 128);
    BlockP b3 = getBlock(128, 256);
    BlockP b4 = getBlock(256, 512);
    Lin summit = getLin(512, 512, true);
    Lin fp4 = getLin(768, 256, true);
    Lin fp3 = getLin(384, 128, true);
    Lin fp2 = getLin(160, 32, true);
    Lin fp1 = getLin(64, 32, true);
    Lin clf1 = getLin(32, 64, true);
    Lin clf2 = getLin(64, 32, true);
    Lin fc = getLin(32, 13, false);

    const int N = 16384, n1 = 4096, n2 = 1024, n3 = 256, n4 = 64;

    Arena all; all.base = (char*)d_ws; all.off = 0; all.cap = ws_size;
    float* h0  = (float*)aalloc(all, (size_t)N  * 32  * 4);
    float* h1  = (float*)aalloc(all, (size_t)N  * 32  * 4);
    float* h2  = (float*)aalloc(all, (size_t)n1 * 128 * 4);
    float* h3  = (float*)aalloc(all, (size_t)n2 * 256 * 4);
    float* h4  = (float*)aalloc(all, (size_t)n3 * 512 * 4);
    float* sm  = (float*)aalloc(all, (size_t)n4 * 512 * 4);
    float* f4  = (float*)aalloc(all, (size_t)n3 * 256 * 4);
    float* f3  = (float*)aalloc(all, (size_t)n2 * 128 * 4);
    float* f2  = (float*)aalloc(all, (size_t)n1 * 32  * 4);
    float* f1  = (float*)aalloc(all, (size_t)N  * 32  * 4);
    float* c1o = (float*)aalloc(all, (size_t)N  * 64  * 4);
    float* c2o = (float*)aalloc(all, (size_t)N  * 32  * 4);
    float* lg  = (float*)aalloc(all, (size_t)N  * 16  * 4);
    Arena scratch = all;  // remaining region, re-used (by value) per stage

    // Encoder (decimation is a contiguous row-prefix: reuse pointers with smaller N)
    h_lin(stream, scratch, x, N, 6, 6, fc0, h0, 0, 0);
    h_block(stream, scratch, h0, 32,  N,  32,  32,  b1, pos, h1);
    h_block(stream, scratch, h1, 32,  n1, 32,  128, b2, pos, h2);
    h_block(stream, scratch, h2, 128, n2, 128, 256, b3, pos, h3);
    h_block(stream, scratch, h3, 256, n3, 256, 512, b4, pos, h4);
    h_lin(stream, scratch, h4, n4, 512, 512, summit, sm, 1, 1);

    // Decoder (feature propagation)
    h_fp(stream, scratch, sm, 512, 512, pos, n4, h3, 256, 256, pos, n3, fp4, f4);
    h_fp(stream, scratch, f4, 256, 256, pos, n3, h2, 128, 128, pos, n2, fp3, f3);
    h_fp(stream, scratch, f3, 128, 128, pos, n2, h1, 32, 32, pos, n1, fp2, f2);
    h_fp(stream, scratch, f2, 32, 32, pos, n1, h1, 32, 32, pos, N, fp1, f1);

    // Classifier head
    h_lin(stream, scratch, f1, N, 32, 32, clf1, c1o, 1, 1);
    h_lin(stream, scratch, c1o, N, 64, 64, clf2, c2o, 1, 1);
    h_lin(stream, scratch, c2o, N, 32, 32, fc, lg, 0, 0);
    k_logsoftmax<<<(N + 255) / 256, 256, 0, stream>>>(lg, 16, N, 13, (float*)d_out);
}
